// AdditiveAttention_29549374997025
// MI455X (gfx1250) — matrix operand order repaired and checked by reading, not compiled, not run on hardware
//
#include <hip/hip_runtime.h>

typedef __attribute__((ext_vector_type(16))) _Float16 v16h;
typedef __attribute__((ext_vector_type(8)))  _Float16 v8h;
typedef __attribute__((ext_vector_type(8)))  float    v8f;
typedef __attribute__((ext_vector_type(2)))  float    v2f;

#define B_  8
#define LQ_ 512
#define LK_ 512
#define D_  256
#define H_  64

// CDNA5 has a native V_TANH_F32 transcendental (ISA trans-op list).
#if __has_builtin(__builtin_amdgcn_tanhf)
  #define FAST_TANH(x) __builtin_amdgcn_tanhf(x)
#elif __has_builtin(__builtin_amdgcn_tanh_f32)
  #define FAST_TANH(x) __builtin_amdgcn_tanh_f32(x)
#else
  #define FAST_TANH(x) tanhf(x)
#endif

// ---------------------------------------------------------------------------
// Kernel 1: qh = Q@Wq, kh = K@Wk  via V_WMMA_F32_16X16X4_F32 (full f32 GEMM).
// One 16x16 output tile per wave, K-loop of 64 steps of K=4.
// ---------------------------------------------------------------------------
__global__ __launch_bounds__(256) void addattn_proj_kernel(
    const float* __restrict__ Q, const float* __restrict__ K,
    const float* __restrict__ Wq, const float* __restrict__ Wk,
    float* __restrict__ qh, float* __restrict__ kh)
{
  const int wave = blockIdx.x * 8 + (threadIdx.x >> 5);  // 2048 waves total
  const int lane = threadIdx.x & 31;
  const int p    = wave >> 10;        // 0 = queries, 1 = keys
  const int rem  = wave & 1023;
  const int m0   = (rem >> 2) * 16;   // 256 M-tiles (4096 rows / 16)
  const int n0   = (rem & 3) * 16;    // 4 N-tiles  (64 cols / 16)

  const float* __restrict__ X = p ? K  : Q;
  const float* __restrict__ W = p ? Wk : Wq;
  float*       __restrict__ dst = p ? kh : qh;

  const int mrow = m0 + (lane & 15);
  const int n    = n0 + (lane & 15);
  const int ksel = (lane < 16) ? 0 : 2;

  v8f c = {};
  for (int k0 = 0; k0 < D_; k0 += 4) {
    const int ka = k0 + ksel;
    v2f a, bvec;
    a.x = X[mrow * D_ + ka];
    a.y = X[mrow * D_ + ka + 1];
    bvec.x = W[ka * H_ + n];
    bvec.y = W[(ka + 1) * H_ + n];
    c = __builtin_amdgcn_wmma_f32_16x16x4_f32(
        false, a, false, bvec, (short)0, c, false, false);
  }

  const int mbase = m0 + ((lane < 16) ? 0 : 8);
#pragma unroll
  for (int i = 0; i < 8; ++i)
    dst[(mbase + i) * H_ + n] = c[i];
}

// ---------------------------------------------------------------------------
// Kernel 2: vt[b][n][k] = (half)values[b][k][n]  (one-time transpose+convert
// so the WMMA B-fragments in the main kernel are contiguous 32B vectors).
// ---------------------------------------------------------------------------
__global__ __launch_bounds__(256) void addattn_vt_kernel(
    const float* __restrict__ values, _Float16* __restrict__ vt)
{
  const int idx = blockIdx.x * 256 + threadIdx.x;   // 1M elements
  const int n = idx & (D_ - 1);
  const int k = (idx >> 8) & (LK_ - 1);
  const int b = idx >> 17;
  vt[(((size_t)b * D_) + n) * LK_ + k] = (_Float16)values[idx];
}

// ---------------------------------------------------------------------------
// Kernel 3: fused scores -> softmax -> (attn @ V).
// One workgroup (512 thr = 16 waves) per (batch, 16-row q tile).
//  Phase 1: thread t owns k-column t; 16 q-rows x 64 h -> 1024 v_tanh/thread.
//  Phase 2: wave w softmaxes score row w with wave32 shfl reductions.
//  Phase 3: wave w computes output N-tile w via V_WMMA_F32_16X16X32_F16.
// ---------------------------------------------------------------------------
__global__ __launch_bounds__(512) void addattn_main_kernel(
    const _Float16* __restrict__ vt, const unsigned char* __restrict__ mask,
    const float* __restrict__ wv,
    const float* __restrict__ qh, const float* __restrict__ kh,
    float* __restrict__ out)
{
  __shared__ float    qh_s[16 * H_];
  __shared__ float    wv_s[H_];
  __shared__ float    S_s[16][LK_];
  __shared__ _Float16 P_s[16][LK_];

  const int b   = blockIdx.x >> 5;   // 8 batches x 32 q-tiles
  const int q0  = (blockIdx.x & 31) * 16;
  const int tid = threadIdx.x;

  for (int i = tid; i < 16 * H_; i += 512)
    qh_s[i] = qh[((size_t)b * LQ_ + q0) * H_ + i];
  if (tid < H_) wv_s[tid] = wv[tid];
  __syncthreads();

  // ---- Phase 1: additive scores with native v_tanh (the compute wall) ----
  {
    const int k = tid;
    const float* __restrict__ khrow = &kh[((size_t)b * LK_ + k) * H_];
    float acc[16];
#pragma unroll
    for (int qi = 0; qi < 16; ++qi) acc[qi] = 0.f;

    for (int h = 0; h < H_; h += 8) {
      float kv[8], wvv[8];
#pragma unroll
      for (int j = 0; j < 8; ++j) { kv[j] = khrow[h + j]; wvv[j] = wv_s[h + j]; }
#pragma unroll
      for (int qi = 0; qi < 16; ++qi) {
#pragma unroll
        for (int j = 0; j < 8; ++j)
          acc[qi] += wvv[j] * FAST_TANH(qh_s[qi * H_ + h + j] + kv[j]);
      }
    }
#pragma unroll
    for (int qi = 0; qi < 16; ++qi) {
      const unsigned char mk = mask[((size_t)b * LQ_ + q0 + qi) * LK_ + k];
      S_s[qi][k] = mk ? -__builtin_inff() : acc[qi];
    }
  }
  __syncthreads();

  // ---- Phase 2: softmax (wave w owns q-row w) ----
  {
    const int w = tid >> 5, lane = tid & 31;
    float vals[16];
    float vmax = -__builtin_inff();
#pragma unroll
    for (int i = 0; i < 16; ++i) {
      vals[i] = S_s[w][lane + 32 * i];
      vmax = fmaxf(vmax, vals[i]);
    }
#pragma unroll
    for (int off = 16; off >= 1; off >>= 1)
      vmax = fmaxf(vmax, __shfl_xor(vmax, off, 32));
    float vsum = 0.f;
#pragma unroll
    for (int i = 0; i < 16; ++i) {
      vals[i] = __expf(vals[i] - vmax);
      vsum += vals[i];
    }
#pragma unroll
    for (int off = 16; off >= 1; off >>= 1)
      vsum += __shfl_xor(vsum, off, 32);
    const float inv = 1.0f / vsum;
#pragma unroll
    for (int i = 0; i < 16; ++i)
      P_s[w][lane + 32 * i] = (_Float16)(vals[i] * inv);
  }
  __syncthreads();

  // ---- Phase 3: O(16x256) = P(16x512) @ V(512x256), wave w -> N-tile w ----
  {
    const int w = tid >> 5, lane = tid & 31;
    const int n = w * 16 + (lane & 15);
    const int q = lane & 15;
    const _Float16* __restrict__ vtc = &vt[(((size_t)b * D_) + n) * LK_];
    v8f c = {};
#pragma unroll 2
    for (int kt = 0; kt < LK_; kt += 32) {
      // A fragment: f16 16x32 layout, two 16B LDS vector loads
      const int abase = kt + ((lane < 16) ? 0 : 16);
      const v8h alo = *(const v8h*)&P_s[q][abase];
      const v8h ahi = *(const v8h*)&P_s[q][abase + 8];
      v16h a;
#pragma unroll
      for (int j = 0; j < 8; ++j) { a[j] = alo[j]; a[8 + j] = ahi[j]; }
      // B fragment: f16 32x16 layout, one 32B global vector load (pre-transposed)
      const int kbase = kt + ((lane < 16) ? 0 : 16);
      const v16h bb = *(const v16h*)(vtc + kbase);
      c = __builtin_amdgcn_wmma_f32_16x16x32_f16(
          false, a, false, bb, (short)0, c, false, false);
    }
    const int mbase = q0 + ((lane < 16) ? 0 : 8);
#pragma unroll
    for (int i = 0; i < 8; ++i)
      out[((size_t)b * LQ_ + mbase + i) * D_ + n] = c[i];
  }
}

// ---------------------------------------------------------------------------
extern "C" void kernel_launch(void* const* d_in, const int* in_sizes, int n_in,
                              void* d_out, int out_size, void* d_ws, size_t ws_size,
                              hipStream_t stream) {
  const float*         Q    = (const float*)d_in[0];
  const float*         K    = (const float*)d_in[1];
  const float*         V    = (const float*)d_in[2];
  const unsigned char* mask = (const unsigned char*)d_in[3];
  const float*         Wq   = (const float*)d_in[4];
  const float*         Wk   = (const float*)d_in[5];
  const float*         wv   = (const float*)d_in[6];
  float* out = (float*)d_out;

  // workspace: qh (1MB) | kh (1MB) | vt (2MB)
  float*    qh = (float*)d_ws;
  float*    kh = qh + (size_t)B_ * LQ_ * H_;
  _Float16* vtp = (_Float16*)(kh + (size_t)B_ * LK_ * H_);

  addattn_proj_kernel<<<256, 256, 0, stream>>>(Q, K, Wq, Wk, qh, kh);
  addattn_vt_kernel<<<(B_ * LK_ * D_) / 256, 256, 0, stream>>>(V, vtp);
  addattn_main_kernel<<<B_ * (LQ_ / 16), 512, 0, stream>>>(vtp, mask, wv, qh, kh, out);
}
